// ed_loss_26680336843532
// MI455X (gfx1250) — compile-verified, matrix operand order checked by reading, not run on hardware
//
#include <hip/hip_runtime.h>

typedef __attribute__((ext_vector_type(16))) __bf16 v16bf;
typedef __attribute__((ext_vector_type(8)))  __bf16 v8bf;
typedef __attribute__((ext_vector_type(4)))  __bf16 v4bf;
typedef __attribute__((ext_vector_type(8)))  float  v8f;

#define NN 16384
#define MM 4096
#define DD 64

// One-time pass: split f32 rows into bf16 (hi, lo) with hi+lo ~= x, and
// compute row squared norms.  Runs once per launch over ~5 MB; removes all
// conversion VALU from the O(N^2) pair loop.
__global__ __launch_bounds__(256) void convert_kernel(
    const float* __restrict__ X, __bf16* __restrict__ hi,
    __bf16* __restrict__ lo, float* __restrict__ norms, int rows) {
  int i = blockIdx.x * blockDim.x + threadIdx.x;
  if (i >= rows) return;
  const float4* p = (const float4*)(X + (size_t)i * DD);
  __bf16* hrow = hi + (size_t)i * DD;
  __bf16* lrow = lo + (size_t)i * DD;
  float s = 0.f;
#pragma unroll
  for (int k = 0; k < DD / 4; ++k) {
    float4 v = p[k];
    float xs[4] = {v.x, v.y, v.z, v.w};
    v4bf hs, ls;
#pragma unroll
    for (int j = 0; j < 4; ++j) {
      float x = xs[j];
      s += x * x;
      __bf16 h = (__bf16)x;
      hs[j] = h;
      ls[j] = (__bf16)(x - (float)h);
    }
    *(v4bf*)(hrow + k * 4) = hs;
    *(v4bf*)(lrow + k * 4) = ls;
  }
  norms[i] = s;
}

// Load one lane's 16 bf16 elements of a 16x16x32 WMMA A/B fragment.
// Row-major source; lane layout per ISA: K = {base..base+7, base+16..base+23},
// base = chunk*32 + (lane>=16 ? 8 : 0), folded into `base` by the caller.
// Two 16-byte loads -> global_load_b128.
__device__ inline v16bf load_frag(const __bf16* __restrict__ rowp, int base) {
  v8bf p0 = *(const v8bf*)(rowp + base);
  v8bf p1 = *(const v8bf*)(rowp + base + 16);
  return __builtin_shufflevector(p0, p1, 0, 1, 2, 3, 4, 5, 6, 7,
                                 8, 9, 10, 11, 12, 13, 14, 15);
}

// 8 waves per WG; wave w owns row-tile blockIdx.x*8+w (16 rows of A) and
// sweeps COLT column tiles.  Per tile: 6 bf16 WMMAs (hi/lo split product,
// 2 K-chunks) + sqrt epilogue.  All WMMA-reaching control flow wave-uniform.
template <bool SELF>
__global__ __launch_bounds__(256) void pairdist_kernel(
    const __bf16* __restrict__ Ahi, const __bf16* __restrict__ Alo,
    const __bf16* __restrict__ Bhi, const __bf16* __restrict__ Blo,
    const float* __restrict__ na, const float* __restrict__ nb,
    float wOff, float wDiag, int colTilesPerWG,
    double* __restrict__ partials) {
  const int wave = threadIdx.x >> 5;
  const int lane = threadIdx.x & 31;
  const int rl   = lane & 15;
  const int hh   = lane >> 4;

  const int rowTile = blockIdx.x * 8 + wave;
  const int rowA0   = rowTile * 16;
  const __bf16* arow_h = Ahi + (size_t)(rowA0 + rl) * DD;
  const __bf16* arow_l = Alo + (size_t)(rowA0 + rl) * DD;

  const v16bf a_hi0 = load_frag(arow_h,  0 + hh * 8);
  const v16bf a_hi1 = load_frag(arow_h, 32 + hh * 8);
  const v16bf a_lo0 = load_frag(arow_l,  0 + hh * 8);
  const v16bf a_lo1 = load_frag(arow_l, 32 + hh * 8);

  float na_reg[8];
#pragma unroll
  for (int r = 0; r < 8; ++r) na_reg[r] = na[rowA0 + hh * 8 + r];

  const int colBase = blockIdx.y * colTilesPerWG;
  int ctStart = 0;
  if (SELF) {
    int d = rowTile - colBase;                 // skip tiles below the diagonal
    ctStart = d > 0 ? (d < colTilesPerWG ? d : colTilesPerWG) : 0;
  }

  float accw = 0.f;
#pragma unroll 2
  for (int ct = ctStart; ct < colTilesPerWG; ++ct) {
    const int colTile = colBase + ct;
    const float w = (SELF && colTile == rowTile) ? wDiag : wOff;

    const __bf16* brow_h = Bhi + (size_t)(colTile * 16 + rl) * DD;
    const __bf16* brow_l = Blo + (size_t)(colTile * 16 + rl) * DD;
    const v16bf b_hi0 = load_frag(brow_h,  0 + hh * 8);
    const v16bf b_hi1 = load_frag(brow_h, 32 + hh * 8);
    const v16bf b_lo0 = load_frag(brow_l,  0 + hh * 8);
    const v16bf b_lo1 = load_frag(brow_l, 32 + hh * 8);

    v8f acc = {0.f, 0.f, 0.f, 0.f, 0.f, 0.f, 0.f, 0.f};
    acc = __builtin_amdgcn_wmma_f32_16x16x32_bf16(false, a_hi0, false, b_hi0, (short)0, acc, false, false);
    acc = __builtin_amdgcn_wmma_f32_16x16x32_bf16(false, a_hi0, false, b_lo0, (short)0, acc, false, false);
    acc = __builtin_amdgcn_wmma_f32_16x16x32_bf16(false, a_lo0, false, b_hi0, (short)0, acc, false, false);
    acc = __builtin_amdgcn_wmma_f32_16x16x32_bf16(false, a_hi1, false, b_hi1, (short)0, acc, false, false);
    acc = __builtin_amdgcn_wmma_f32_16x16x32_bf16(false, a_hi1, false, b_lo1, (short)0, acc, false, false);
    acc = __builtin_amdgcn_wmma_f32_16x16x32_bf16(false, a_lo1, false, b_hi1, (short)0, acc, false, false);

    const float nbv = nb[colTile * 16 + rl];
    float ts = 0.f;
#pragma unroll
    for (int r = 0; r < 8; ++r) {
      // C layout: VGPR r -> M = r (+8 for lanes 16-31); N = lane & 15
      float sq = fmaf(-2.f, acc[r], na_reg[r] + nbv);
      sq = fmaxf(sq, 0.f);
      ts += __builtin_amdgcn_sqrtf(sq);
    }
    accw = fmaf(w, ts, accw);
  }

#pragma unroll
  for (int off = 16; off >= 1; off >>= 1) accw += __shfl_xor(accw, off, 32);

  __shared__ float wsum[8];
  if (lane == 0) wsum[wave] = accw;
  __syncthreads();
  if (threadIdx.x == 0) {
    float s = 0.f;
#pragma unroll
    for (int i = 0; i < 8; ++i) s += wsum[i];
    partials[(size_t)blockIdx.y * gridDim.x + blockIdx.x] = (double)s;
  }
}

__global__ __launch_bounds__(256) void final_reduce_kernel(
    const double* __restrict__ partials, int count, float* __restrict__ out) {
  __shared__ double sh[256];
  double s = 0.0;
  for (int i = threadIdx.x; i < count; i += 256) s += partials[i];
  sh[threadIdx.x] = s;
  __syncthreads();
  for (int off = 128; off >= 1; off >>= 1) {
    if (threadIdx.x < off) sh[threadIdx.x] += sh[threadIdx.x + off];
    __syncthreads();
  }
  if (threadIdx.x == 0) out[0] = (float)sh[0];
}

extern "C" void kernel_launch(void* const* d_in, const int* in_sizes, int n_in,
                              void* d_out, int out_size, void* d_ws, size_t ws_size,
                              hipStream_t stream) {
  (void)in_sizes; (void)n_in; (void)out_size; (void)ws_size;
  const float* latent = (const float*)d_in[0];  // (16384, 64) f32
  const float* z      = (const float*)d_in[1];  // (4096, 64)  f32
  float* out = (float*)d_out;

  char* ws = (char*)d_ws;
  size_t off = 0;
  __bf16* lat_hi = (__bf16*)(ws + off); off += (size_t)NN * DD * 2;  // 2 MB
  __bf16* lat_lo = (__bf16*)(ws + off); off += (size_t)NN * DD * 2;  // 2 MB
  __bf16* z_hi   = (__bf16*)(ws + off); off += (size_t)MM * DD * 2;  // 0.5 MB
  __bf16* z_lo   = (__bf16*)(ws + off); off += (size_t)MM * DD * 2;  // 0.5 MB
  float*  na     = (float*)(ws + off);  off += (size_t)NN * 4;
  float*  nb     = (float*)(ws + off);  off += (size_t)MM * 4;
  double* partials = (double*)(ws + off);                            // 8B aligned

  convert_kernel<<<(NN + 255) / 256, 256, 0, stream>>>(latent, lat_hi, lat_lo, na, NN);
  convert_kernel<<<(MM + 255) / 256, 256, 0, stream>>>(z, z_hi, z_lo, nb, MM);

  const double invD = 1.0 / (double)DD;
  const float w_lz = (float)( 2.0 / ((double)NN * (double)MM) * invD);
  const float w_ll = (float)(-1.0 / ((double)NN * (double)NN) * invD);
  const float w_zz = (float)(-1.0 / ((double)MM * (double)MM) * invD);

  const int COLT = 16;

  dim3 g0(NN / 128, (MM / 16) / COLT);   // 128 x 16  = 2048 WGs
  pairdist_kernel<false><<<g0, 256, 0, stream>>>(
      lat_hi, lat_lo, z_hi, z_lo, na, nb, w_lz, w_lz, COLT, partials);
  int p0 = g0.x * g0.y;

  dim3 g1(NN / 128, (NN / 16) / COLT);   // 128 x 64  = 8192 WGs
  pairdist_kernel<true><<<g1, 256, 0, stream>>>(
      lat_hi, lat_lo, lat_hi, lat_lo, na, na, 2.f * w_ll, w_ll, COLT, partials + p0);
  int p1 = g1.x * g1.y;

  dim3 g2(MM / 128, (MM / 16) / COLT);   // 32 x 16   = 512 WGs
  pairdist_kernel<true><<<g2, 256, 0, stream>>>(
      z_hi, z_lo, z_hi, z_lo, nb, nb, 2.f * w_zz, w_zz, COLT, partials + p0 + p1);
  int total = p0 + p1 + g2.x * g2.y;

  final_reduce_kernel<<<1, 256, 0, stream>>>(partials, total, out);
}
